// RNNDecoder_77970836291617
// MI455X (gfx1250) — matrix operand order repaired and checked by reading, not compiled, not run on hardware
//
#include <hip/hip_runtime.h>
#include <hip/hip_bf16.h>

// ---------------- types ----------------
typedef __bf16 bf16_t;
typedef __attribute__((ext_vector_type(16))) __bf16 v16bf;
typedef __attribute__((ext_vector_type(8)))  __bf16 v8bf;
typedef __attribute__((ext_vector_type(8)))  float  v8f;

#define B_  32
#define S_  128
#define H_  512
#define E_  512
#define V_  32000
#define G4H 2048          // 4*H
#define NROWS 4096        // S*B
#define NVT 8             // v-tiles in projection
#define VT  4000          // cols per v-tile (8*4000 = 32000)

__device__ __forceinline__ v8f wmma_bf16(v16bf a, v16bf b, v8f c) {
  return __builtin_amdgcn_wmma_f32_16x16x32_bf16(false, a, false, b, (short)0, c, false, false);
}

// A fragment: 16x32 bf16, row-major source, `base` already points at this lane's row.
// ISA 7.12.2: elems 0..7 -> K = khalf*8 + e ; elems 8..15 -> K = 16 + khalf*8 + (e-8)
__device__ __forceinline__ v16bf load_frag_a(const bf16_t* base, int k0, int lane) {
  int khalf = (lane >> 4) & 1;
  union { v16bf v; v8bf h[2]; } u;
  u.h[0] = *(const v8bf*)(base + k0 + khalf * 8);
  u.h[1] = *(const v8bf*)(base + k0 + 16 + khalf * 8);
  return u.v;
}

// B fragment: 32x16 bf16; `wrow` points at W[col,:] (row-major [N,K], so B[k,n]=W[n,k]).
// Same K order as A: elems 0..7 -> K = khalf*8 + e ; elems 8..15 -> K = 16 + khalf*8 + (e-8)
__device__ __forceinline__ v16bf load_frag_b(const bf16_t* wrow, int k0, int lane) {
  int kh = ((lane >> 4) & 1) * 8;
  union { v16bf v; v8bf h[2]; } u;
  u.h[0] = *(const v8bf*)(wrow + k0 + kh);
  u.h[1] = *(const v8bf*)(wrow + k0 + 16 + kh);
  return u.v;
}

// ---------------- kernel 0: convert weights to bf16, build bias/state ----------------
#define NWIH  (G4H * E_)
#define NWHH  (G4H * H_)
#define NWOUT ((long)V_ * H_)
#define NEMB  ((long)V_ * E_)
__global__ __launch_bounds__(256) void convert_all(
    const float* Wih, const float* Whh, const float* Wout, const float* emb,
    const float* bih, const float* bhh, const float* h0, const float* c0,
    bf16_t* Wih_bf, bf16_t* Whh_bf, bf16_t* Wout_bf, bf16_t* emb_bf,
    float* gate_bias, bf16_t* hprev0, float* c_ws) {
  const long TOTAL = (long)NWIH + NWHH + NWOUT + NEMB + G4H + B_ * H_ + B_ * H_;
  long stride = (long)gridDim.x * blockDim.x;
  for (long i = (long)blockIdx.x * blockDim.x + threadIdx.x; i < TOTAL; i += stride) {
    long j = i;
    if (j < NWIH)  { Wih_bf[j]  = (bf16_t)Wih[j];  continue; } j -= NWIH;
    if (j < NWHH)  { Whh_bf[j]  = (bf16_t)Whh[j];  continue; } j -= NWHH;
    if (j < NWOUT) { Wout_bf[j] = (bf16_t)Wout[j]; continue; } j -= NWOUT;
    if (j < NEMB)  { emb_bf[j]  = (bf16_t)emb[j];  continue; } j -= NEMB;
    if (j < G4H)   { gate_bias[j] = bih[j] + bhh[j]; continue; } j -= G4H;
    if (j < B_*H_) { hprev0[j] = (bf16_t)h0[j]; continue; } j -= B_*H_;
    c_ws[j] = c0[j];
  }
}

// ---------------- kernel 1: Gx = emb[tok] @ W_ih^T + (b_ih+b_hh)  [4096 x 2048] ----------------
__global__ __launch_bounds__(256) void gx_gemm(
    const bf16_t* __restrict__ emb_bf, const bf16_t* __restrict__ Wih_bf,
    const float* __restrict__ gate_bias, const int* __restrict__ toks,
    float* __restrict__ Gx) {
  int wave = threadIdx.x >> 5, lane = threadIdx.x & 31;
  int mb = wave & 1, nb = wave >> 1;                 // 2 M-tiles x 4 N-tiles
  int rowbase = blockIdx.y * 32 + mb * 16;
  int colbase = blockIdx.x * 64 + nb * 16;
  int lrow = rowbase + (lane & 15);                  // R = s*32 + b
  int s = lrow >> 5, b = lrow & 31;
  int tok = toks[b * S_ + s];
  const bf16_t* arow = emb_bf + (long)tok * E_;
  const bf16_t* brow = Wih_bf + (long)(colbase + (lane & 15)) * E_;
  v8f acc = {0.f,0.f,0.f,0.f,0.f,0.f,0.f,0.f};
#pragma unroll 4
  for (int k0 = 0; k0 < E_; k0 += 32) {
    v16bf a = load_frag_a(arow, k0, lane);
    v16bf bb = load_frag_b(brow, k0, lane);
    acc = wmma_bf16(a, bb, acc);
  }
  float bias = gate_bias[colbase + (lane & 15)];
  int rr = rowbase + ((lane >> 4) << 3);
  int cc = colbase + (lane & 15);
#pragma unroll
  for (int r = 0; r < 8; ++r)
    Gx[(long)(rr + r) * G4H + cc] = acc[r] + bias;
}

// ---------------- kernel 2 (x128): one LSTM step ----------------
// gates = Gx[t] + h_prev @ W_hh^T ; cell update ; h -> bf16 into Hbf[t]
__global__ __launch_bounds__(256) void lstm_step(
    const bf16_t* __restrict__ hprev_bf, const bf16_t* __restrict__ Whh_bf,
    const float* __restrict__ Gx_t, float* __restrict__ c_st,
    bf16_t* __restrict__ hout_bf) {
  __shared__ float gbuf[4][32][64];
  int wave = threadIdx.x >> 5, lane = threadIdx.x & 31;
  int g = wave >> 1, m = wave & 1;
  int hbase = blockIdx.x * 64;
  int rowb = m * 16;
  const bf16_t* arow = hprev_bf + (long)(rowb + (lane & 15)) * H_;
  v8f acc[4];
#pragma unroll
  for (int nt = 0; nt < 4; ++nt) acc[nt] = (v8f){0.f,0.f,0.f,0.f,0.f,0.f,0.f,0.f};
  const bf16_t* brow[4];
#pragma unroll
  for (int nt = 0; nt < 4; ++nt)
    brow[nt] = Whh_bf + (long)(g * H_ * 1 + 0) * 0 + (long)(g * 512 + hbase + nt * 16 + (lane & 15)) * H_;
  for (int k0 = 0; k0 < H_; k0 += 32) {
    v16bf a = load_frag_a(arow, k0, lane);
#pragma unroll
    for (int nt = 0; nt < 4; ++nt) {
      v16bf bb = load_frag_b(brow[nt], k0, lane);
      acc[nt] = wmma_bf16(a, bb, acc[nt]);
    }
  }
  int rr = rowb + ((lane >> 4) << 3);
#pragma unroll
  for (int nt = 0; nt < 4; ++nt)
#pragma unroll
    for (int r = 0; r < 8; ++r)
      gbuf[g][rr + r][nt * 16 + (lane & 15)] = acc[nt][r];
  __syncthreads();
  // elementwise cell update: 32 rows x 64 h-cols, 8 per thread
  int row = threadIdx.x >> 3;
  int h0i = (threadIdx.x & 7) * 8;
#pragma unroll
  for (int j = 0; j < 8; ++j) {
    int hh = h0i + j;
    int col = hbase + hh;
    const float* gx = Gx_t + (long)row * G4H;
    float iv = gbuf[0][row][hh] + gx[col];
    float fv = gbuf[1][row][hh] + gx[512 + col];
    float gv = gbuf[2][row][hh] + gx[1024 + col];
    float ov = gbuf[3][row][hh] + gx[1536 + col];
    iv = 1.0f / (1.0f + __expf(-iv));
    fv = 1.0f / (1.0f + __expf(-fv));
    gv = tanhf(gv);
    ov = 1.0f / (1.0f + __expf(-ov));
    long ci = (long)row * H_ + col;
    float c = fv * c_st[ci] + iv * gv;
    c_st[ci] = c;
    float h = ov * tanhf(c);
    hout_bf[ci] = (bf16_t)h;
  }
}

// ---------------- kernel 3: projection partials with online softmax ----------------
// grid (NVT, 64): each WG does 64 rows x 4000 cols streaming; partials per (vtile*2+nh, row)
__global__ __launch_bounds__(256) void proj_partial(
    const bf16_t* __restrict__ Hbf, const bf16_t* __restrict__ Wout_bf,
    const float* __restrict__ b_out,
    float* __restrict__ pmax, float* __restrict__ psum, int* __restrict__ pidx) {
  __shared__ bf16_t aswz[4 * 16 * 2 * 32 * 8];      // 64 KB, fragment-swizzled A (64x512 bf16)
  int rowbase = blockIdx.y * 64;
  int vt = blockIdx.x;
  int n_begin = vt * VT;
  // cooperative load of A block, scattered into per-fragment 16B chunks (conflict-free reads)
  for (int i = threadIdx.x; i < 64 * 64; i += 256) {
    int r = i >> 6;                                  // local row 0..63
    int gch = i & 63;                                // 8-elem K group
    v8bf d = *(const v8bf*)(Hbf + (long)(rowbase + r) * H_ + gch * 8);
    int mb = r >> 4, Lr = r & 15;
    int k0i = gch >> 2;
    int c = (gch >> 1) & 1;
    int khalf = gch & 1;
    int lanei = khalf * 16 + Lr;
    int off = (((mb * 16 + k0i) * 2 + c) * 32 + lanei) * 8;
    *(v8bf*)(aswz + off) = d;
  }
  __syncthreads();
  int wave = threadIdx.x >> 5, lane = threadIdx.x & 31;
  int mb = wave >> 1, nh = wave & 1;                 // 4 M-tiles x 2 N-halves
  const bf16_t* abase = aswz + mb * (16 * 2 * 32 * 8);
  float mx[8]; float sm[8]; int ax[8];
#pragma unroll
  for (int r = 0; r < 8; ++r) { mx[r] = -INFINITY; sm[r] = 0.f; ax[r] = 0; }
  for (int n0 = 0; n0 < VT; n0 += 32) {
    int col = n_begin + n0 + nh * 16 + (lane & 15);
    const bf16_t* brow = Wout_bf + (long)col * H_;
    v8f acc = {0.f,0.f,0.f,0.f,0.f,0.f,0.f,0.f};
#pragma unroll 4
    for (int k0i = 0; k0i < 16; ++k0i) {
      union { v16bf v; v8bf h[2]; } ua;
      ua.h[0] = *(const v8bf*)(abase + ((k0i * 2 + 0) * 32 + lane) * 8);
      ua.h[1] = *(const v8bf*)(abase + ((k0i * 2 + 1) * 32 + lane) * 8);
      v16bf bb = load_frag_b(brow, k0i * 32, lane);
      acc = wmma_bf16(ua.v, bb, acc);
    }
    float bias = b_out[col];
#pragma unroll
    for (int r = 0; r < 8; ++r) {
      float x = acc[r] + bias;
      float nm = fmaxf(mx[r], x);
      sm[r] = sm[r] * __expf(mx[r] - nm) + __expf(x - nm);
      ax[r] = (x > mx[r]) ? col : ax[r];
      mx[r] = nm;
    }
  }
  // reduce across 16 lanes (same row, different cols)
#pragma unroll
  for (int msk = 1; msk <= 8; msk <<= 1) {
#pragma unroll
    for (int r = 0; r < 8; ++r) {
      float om = __shfl_xor(mx[r], msk, 16);
      float os = __shfl_xor(sm[r], msk, 16);
      int   oa = __shfl_xor(ax[r], msk, 16);
      float nm = fmaxf(mx[r], om);
      float s  = sm[r] * __expf(mx[r] - nm) + os * __expf(om - nm);
      int a;
      if (om > mx[r]) a = oa;
      else if (mx[r] > om) a = ax[r];
      else a = (ax[r] < oa) ? ax[r] : oa;          // tie: first occurrence
      mx[r] = nm; sm[r] = s; ax[r] = a;
    }
  }
  if ((lane & 15) == 0) {
    int pv = vt * 2 + nh;
#pragma unroll
    for (int r = 0; r < 8; ++r) {
      int row = rowbase + mb * 16 + r + ((lane >> 4) << 3);
      long idx = (long)pv * NROWS + row;
      pmax[idx] = mx[r]; psum[idx] = sm[r]; pidx[idx] = ax[r];
    }
  }
}

// ---------------- kernel 4: combine partials, token logit, nll + argmax ----------------
__global__ __launch_bounds__(256) void combine(
    const float* __restrict__ pmax, const float* __restrict__ psum, const int* __restrict__ pidx,
    const bf16_t* __restrict__ Hbf, const bf16_t* __restrict__ Wout_bf,
    const float* __restrict__ b_out, const int* __restrict__ toks,
    float* __restrict__ nll_arr, float* __restrict__ out_sym) {
  int wave = threadIdx.x >> 5, lane = threadIdx.x & 31;
  int row = blockIdx.x * 8 + wave;
  int pv = lane & 15;
  long idx = (long)pv * NROWS + row;
  float mx = pmax[idx];
  float sm = psum[idx];
  int   ax = pidx[idx];
#pragma unroll
  for (int msk = 1; msk <= 8; msk <<= 1) {
    float om = __shfl_xor(mx, msk, 16);
    float os = __shfl_xor(sm, msk, 16);
    int   oa = __shfl_xor(ax, msk, 16);
    float nm = fmaxf(mx, om);
    float s  = sm * __expf(mx - nm) + os * __expf(om - nm);
    int a;
    if (om > mx) a = oa;
    else if (mx > om) a = ax;
    else a = (ax < oa) ? ax : oa;
    mx = nm; sm = s; ax = a;
  }
  float MX = __shfl(mx, 0);
  float SM = __shfl(sm, 0);
  int   AX = __shfl(ax, 0);
  float logZ = MX + __logf(SM);
  int s = row >> 5, b = row & 31;
  int tok = toks[b * S_ + s];
  const bf16_t* hrow = Hbf + (long)row * H_;
  const bf16_t* wrow = Wout_bf + (long)tok * H_;
  float d = 0.f;
  {
    const v8bf* hp = (const v8bf*)(hrow + lane * 16);
    const v8bf* wp = (const v8bf*)(wrow + lane * 16);
    v8bf h0v = hp[0], h1v = hp[1], w0v = wp[0], w1v = wp[1];
#pragma unroll
    for (int e = 0; e < 8; ++e) d += (float)h0v[e] * (float)w0v[e] + (float)h1v[e] * (float)w1v[e];
  }
#pragma unroll
  for (int msk = 1; msk < 32; msk <<= 1) d += __shfl_xor(d, msk, 32);
  if (lane == 0) {
    float logit = d + b_out[tok];
    nll_arr[row] = (logZ - logit) * (1.0f / (float)B_);
    out_sym[b * S_ + s] = (float)AX;
  }
}

// ---------------- kernel 5: reduce loss ----------------
__global__ __launch_bounds__(256) void final_reduce(const float* __restrict__ nll_arr,
                                                    float* __restrict__ out0) {
  __shared__ float red[256];
  float s = 0.f;
  for (int i = threadIdx.x; i < NROWS; i += 256) s += nll_arr[i];
  red[threadIdx.x] = s;
  __syncthreads();
  for (int off = 128; off > 0; off >>= 1) {
    if (threadIdx.x < off) red[threadIdx.x] += red[threadIdx.x + off];
    __syncthreads();
  }
  if (threadIdx.x == 0) out0[0] = red[0];
}

// ---------------- launch ----------------
extern "C" void kernel_launch(void* const* d_in, const int* in_sizes, int n_in,
                              void* d_out, int out_size, void* d_ws, size_t ws_size,
                              hipStream_t stream) {
  const int*   toks  = (const int*)d_in[0];
  const float* h0    = (const float*)d_in[3];
  const float* c0    = (const float*)d_in[4];
  const float* emb   = (const float*)d_in[5];
  const float* W_ih  = (const float*)d_in[6];
  const float* W_hh  = (const float*)d_in[7];
  const float* b_ih  = (const float*)d_in[8];
  const float* b_hh  = (const float*)d_in[9];
  const float* W_out = (const float*)d_in[10];
  const float* b_out = (const float*)d_in[11];
  float* out = (float*)d_out;

  // carve workspace
  char* base = (char*)d_ws;
  size_t off = 0;
  auto carve = [&](size_t bytes) -> void* {
    off = (off + 255) & ~(size_t)255;
    void* r = base + off;
    off += bytes;
    return r;
  };
  bf16_t* emb_bf  = (bf16_t*)carve(NEMB * 2);
  bf16_t* Wout_bf = (bf16_t*)carve(NWOUT * 2);
  bf16_t* Wih_bf  = (bf16_t*)carve((size_t)NWIH * 2);
  bf16_t* Whh_bf  = (bf16_t*)carve((size_t)NWHH * 2);
  float*  Gx      = (float*)carve((size_t)NROWS * G4H * 4);
  bf16_t* Hbf     = (bf16_t*)carve((size_t)NROWS * H_ * 2);
  float*  gbias   = (float*)carve(G4H * 4);
  bf16_t* hprev0  = (bf16_t*)carve((size_t)B_ * H_ * 2);
  float*  c_ws    = (float*)carve((size_t)B_ * H_ * 4);
  float*  pmax    = (float*)carve((size_t)16 * NROWS * 4);
  float*  psum    = (float*)carve((size_t)16 * NROWS * 4);
  int*    pidx    = (int*)carve((size_t)16 * NROWS * 4);
  float*  nll     = (float*)carve((size_t)NROWS * 4);

  convert_all<<<1024, 256, 0, stream>>>(W_ih, W_hh, W_out, emb, b_ih, b_hh, h0, c0,
                                        Wih_bf, Whh_bf, Wout_bf, emb_bf,
                                        gbias, hprev0, c_ws);

  gx_gemm<<<dim3(G4H / 64, NROWS / 32), 256, 0, stream>>>(emb_bf, Wih_bf, gbias, toks, Gx);

  for (int t = 0; t < S_; ++t) {
    const bf16_t* hprev = (t == 0) ? hprev0 : (Hbf + (long)(t - 1) * B_ * H_);
    lstm_step<<<H_ / 64, 256, 0, stream>>>(hprev, Whh_bf,
                                           Gx + (long)t * B_ * G4H, c_ws,
                                           Hbf + (long)t * B_ * H_);
  }

  proj_partial<<<dim3(NVT, NROWS / 64), 256, 0, stream>>>(Hbf, Wout_bf, b_out,
                                                          pmax, psum, pidx);

  combine<<<NROWS / 8, 256, 0, stream>>>(pmax, psum, pidx, Hbf, Wout_bf, b_out, toks,
                                         nll, out + 1);

  final_reduce<<<1, 256, 0, stream>>>(nll, out);
}
